// GATDefectLocalizer_30382598652511
// MI455X (gfx1250) — matrix operand order repaired and checked by reading, not compiled, not run on hardware
//
#include <hip/hip_runtime.h>
#include <hip/hip_bf16.h>

typedef __attribute__((ext_vector_type(16))) __bf16 v16bf;
typedef __attribute__((ext_vector_type(8)))  float  v8f;

#define N_NODES 50000
#define HID 64
#define F1 256   // HEADS*HID
#define NEG_SLOPE 0.2f

__device__ __forceinline__ unsigned fenc(float f) {
  unsigned u = __float_as_uint(f);
  return (u & 0x80000000u) ? ~u : (u | 0x80000000u);
}
__device__ __forceinline__ float fdec(unsigned u) {
  return __uint_as_float((u & 0x80000000u) ? (u & 0x7fffffffu) : ~u);
}
__device__ __forceinline__ float lrelu(float v) { return v > 0.f ? v : NEG_SLOPE * v; }
__device__ __forceinline__ float eluf(float v)  { return v > 0.f ? v : __expf(v) - 1.f; }

// ---------------- zero fill ----------------
__global__ void k_fill0(unsigned* __restrict__ p, long n) {
  long i = (long)blockIdx.x * blockDim.x + threadIdx.x;
  if (i < n) p[i] = 0u;
}

// ---------------- layer 1: x@W1 (K=3) + attention halves; wave per node ----------------
__global__ void k_gemm1_alpha(const float* __restrict__ x, const float* __restrict__ W1,
                              const float* __restrict__ as1, const float* __restrict__ ad1,
                              float* __restrict__ h1, float* __restrict__ alpha_s,
                              float* __restrict__ alpha_d) {
  int wid = threadIdx.x >> 5, lane = threadIdx.x & 31;
  int node = blockIdx.x * (blockDim.x >> 5) + wid;
  if (node >= N_NODES) return;
  float x0 = x[node*3+0], x1 = x[node*3+1], x2 = x[node*3+2];
  float sa[4] = {0.f,0.f,0.f,0.f}, sd[4] = {0.f,0.f,0.f,0.f};
#pragma unroll
  for (int j = 0; j < 8; ++j) {
    int c = lane + 32*j;                 // coalesced channel
    float hc = x0*W1[c] + x1*W1[F1+c] + x2*W1[2*F1+c];
    h1[(long)node*F1 + c] = hc;
    int hd = j >> 1;                     // head = c>>6, uniform per j
    int cc = c & 63;
    sa[hd] += hc * as1[hd*64+cc];
    sd[hd] += hc * ad1[hd*64+cc];
  }
#pragma unroll
  for (int off = 16; off > 0; off >>= 1) {
#pragma unroll
    for (int hd = 0; hd < 4; ++hd) {
      sa[hd] += __shfl_xor(sa[hd], off, 32);
      sd[hd] += __shfl_xor(sd[hd], off, 32);
    }
  }
  if (lane == 0) {
#pragma unroll
    for (int hd = 0; hd < 4; ++hd) {
      alpha_s[node*4+hd] = sa[hd];
      alpha_d[node*4+hd] = sd[hd];
    }
  }
}

// ---------------- W2 -> bf16, transposed to [64][256] ----------------
__global__ void k_w2t(const float* __restrict__ W2, __bf16* __restrict__ w2t) {
  int i = blockIdx.x * blockDim.x + threadIdx.x;
  if (i >= 256*64) return;
  int c = i & 63, k = i >> 6;
  w2t[c*256 + k] = (__bf16)W2[i];
}

// ---------------- layer 1 edge pass A: segment max (encoded atomicMax) ----------------
__global__ void k_edge_max1(const int* __restrict__ ei, const float* __restrict__ as,
                            const float* __restrict__ ad, unsigned* __restrict__ emax,
                            int E, int ET) {
  int e = blockIdx.x * blockDim.x + threadIdx.x;
  if (e >= ET) return;
  int s, d;
  if (e < E) { s = ei[e]; d = ei[E + e]; } else { s = d = e - E; }
#pragma unroll
  for (int hd = 0; hd < 4; ++hd) {
    float v = lrelu(as[s*4+hd] + ad[d*4+hd]);
    atomicMax(&emax[d*4+hd], fenc(v));
  }
}

// ---------------- layer 1 edge pass B: exp, denom, weighted scatter; wave per edge ----------------
__global__ void k_edge_agg1(const int* __restrict__ ei, const float* __restrict__ as,
                            const float* __restrict__ ad, const unsigned* __restrict__ emax,
                            float* __restrict__ denom, const float* __restrict__ h1,
                            float* __restrict__ out1, int E, int ET) {
  int wid = threadIdx.x >> 5, lane = threadIdx.x & 31;
  int e = blockIdx.x * (blockDim.x >> 5) + wid;
  if (e >= ET) return;
  int s, d;
  if (e < E) { s = ei[e]; d = ei[E + e]; } else { s = d = e - E; }
  float ex[4];
#pragma unroll
  for (int hd = 0; hd < 4; ++hd) {
    float v = lrelu(as[s*4+hd] + ad[d*4+hd]);
    ex[hd] = __expf(v - fdec(emax[d*4+hd]));
  }
  if (lane < 4) {
    float v = lrelu(as[s*4+lane] + ad[d*4+lane]);
    atomicAdd(&denom[d*4+lane], __expf(v - fdec(emax[d*4+lane])));
  }
#pragma unroll
  for (int j = 0; j < 8; ++j) {
    int c = lane + 32*j;
    atomicAdd(&out1[(long)d*F1 + c], ex[j>>1] * h1[(long)s*F1 + c]);
  }
}

// ---------------- normalize + bias + ELU -> bf16 (A matrix for WMMA) ----------------
__global__ void k_norm1(const float* __restrict__ out1, const float* __restrict__ denom,
                        const float* __restrict__ b1, __bf16* __restrict__ hb) {
  long i = (long)blockIdx.x * blockDim.x + threadIdx.x;
  if (i >= (long)N_NODES * F1) return;
  int c = (int)(i & (F1 - 1));
  int n = (int)(i >> 8);
  float v = out1[i] / (denom[n*4 + (c>>6)] + 1e-16f) + b1[c];
  hb[i] = (__bf16)eluf(v);
}

// ---------------- layer 2 GEMM: [N,256]bf16 @ [256,64]bf16 -> [N,64]f32 via WMMA ----------------
__global__ void __launch_bounds__(128)
k_gemm2_wmma(const __bf16* __restrict__ hb, const __bf16* __restrict__ w2t,
             float* __restrict__ h2) {
  int lane  = threadIdx.x & 31;
  int wtile = blockIdx.x * (blockDim.x >> 5) + (threadIdx.x >> 5);
  if (wtile >= N_NODES/16) return;                 // 50000 = 3125*16 exactly
  int m  = lane & 15;                              // row (A) / col (B,C)
  int hf = lane >> 4;                              // lane half
  const __bf16* Arow = hb + (long)(wtile*16 + m) * F1;
  v8f acc[4] = {};
  for (int s = 0; s < 8; ++s) {                    // K = 256 in 8 steps of 32
    int kb = s * 32;
    v16bf a;
    // A 16x32 bf16 layout: v -> K = hf*16 + 2v,2v+1
#pragma unroll
    for (int v = 0; v < 8; ++v) {
      int ko = hf*16 + 2*v;
      a[2*v]   = Arow[kb + ko];
      a[2*v+1] = Arow[kb + ko + 1];
    }
#pragma unroll
    for (int g = 0; g < 4; ++g) {                  // 4 N-tiles of 16 cols
      const __bf16* Bcol = w2t + (long)(g*16 + m) * F1;
      v16bf b;
      // B 32x16 layout: lanes 0-15 hold K=0..15, lanes 16-31 hold K=16..31; v -> K=2v,2v+1
#pragma unroll
      for (int v = 0; v < 8; ++v) {
        int ko = hf*16 + 2*v;
        b[2*v]   = Bcol[kb + ko];
        b[2*v+1] = Bcol[kb + ko + 1];
      }
      acc[g] = __builtin_amdgcn_wmma_f32_16x16x32_bf16(
          false, a, false, b, (short)0, acc[g], false, false);
    }
  }
  // C layout: VGPR r -> M = hf*8 + r, N = m
#pragma unroll
  for (int g = 0; g < 4; ++g)
#pragma unroll
    for (int r = 0; r < 8; ++r)
      h2[(long)(wtile*16 + hf*8 + r) * HID + g*16 + m] = acc[g][r];
}

// ---------------- layer 2 attention halves; wave per node ----------------
__global__ void k_alpha2(const float* __restrict__ h2, const float* __restrict__ as2,
                         const float* __restrict__ ad2, float* __restrict__ alpha_s,
                         float* __restrict__ alpha_d) {
  int wid = threadIdx.x >> 5, lane = threadIdx.x & 31;
  int node = blockIdx.x * (blockDim.x >> 5) + wid;
  if (node >= N_NODES) return;
  float v0 = h2[(long)node*64 + lane], v1 = h2[(long)node*64 + lane + 32];
  float sa = v0*as2[lane] + v1*as2[lane+32];
  float sd = v0*ad2[lane] + v1*ad2[lane+32];
#pragma unroll
  for (int off = 16; off > 0; off >>= 1) {
    sa += __shfl_xor(sa, off, 32);
    sd += __shfl_xor(sd, off, 32);
  }
  if (lane == 0) { alpha_s[node] = sa; alpha_d[node] = sd; }
}

// ---------------- layer 2 edge passes ----------------
__global__ void k_edge_max2(const int* __restrict__ ei, const float* __restrict__ as,
                            const float* __restrict__ ad, unsigned* __restrict__ emax,
                            int E, int ET) {
  int e = blockIdx.x * blockDim.x + threadIdx.x;
  if (e >= ET) return;
  int s, d;
  if (e < E) { s = ei[e]; d = ei[E + e]; } else { s = d = e - E; }
  atomicMax(&emax[d], fenc(lrelu(as[s] + ad[d])));
}

__global__ void k_edge_agg2(const int* __restrict__ ei, const float* __restrict__ as,
                            const float* __restrict__ ad, const unsigned* __restrict__ emax,
                            float* __restrict__ denom, const float* __restrict__ h2,
                            float* __restrict__ out2, int E, int ET) {
  int wid = threadIdx.x >> 5, lane = threadIdx.x & 31;
  int e = blockIdx.x * (blockDim.x >> 5) + wid;
  if (e >= ET) return;
  int s, d;
  if (e < E) { s = ei[e]; d = ei[E + e]; } else { s = d = e - E; }
  float ex = __expf(lrelu(as[s] + ad[d]) - fdec(emax[d]));
  if (lane == 0) atomicAdd(&denom[d], ex);
  atomicAdd(&out2[(long)d*64 + lane],      ex * h2[(long)s*64 + lane]);
  atomicAdd(&out2[(long)d*64 + lane + 32], ex * h2[(long)s*64 + lane + 32]);
}

// ---------------- finalize: normalize + b2 + ELU + FC + sigmoid; wave per node ----------------
__global__ void k_final(const float* __restrict__ out2, const float* __restrict__ denom,
                        const float* __restrict__ b2, const float* __restrict__ Wfc,
                        const float* __restrict__ bfc, float* __restrict__ out) {
  int wid = threadIdx.x >> 5, lane = threadIdx.x & 31;
  int node = blockIdx.x * (blockDim.x >> 5) + wid;
  if (node >= N_NODES) return;
  float inv = 1.f / (denom[node] + 1e-16f);
  float p = 0.f;
#pragma unroll
  for (int j = 0; j < 2; ++j) {
    int c = lane + 32*j;
    float v = out2[(long)node*64 + c] * inv + b2[c];
    p += eluf(v) * Wfc[c];
  }
#pragma unroll
  for (int off = 16; off > 0; off >>= 1) p += __shfl_xor(p, off, 32);
  if (lane == 0) out[node] = 1.f / (1.f + __expf(-(p + bfc[0])));
}

extern "C" void kernel_launch(void* const* d_in, const int* in_sizes, int n_in,
                              void* d_out, int out_size, void* d_ws, size_t ws_size,
                              hipStream_t stream) {
  (void)n_in; (void)out_size; (void)ws_size;
  const float* x   = (const float*)d_in[0];
  const int*   ei  = (const int*)d_in[1];
  const float* W1  = (const float*)d_in[2];
  const float* as1 = (const float*)d_in[3];
  const float* ad1 = (const float*)d_in[4];
  const float* b1  = (const float*)d_in[5];
  const float* W2  = (const float*)d_in[6];
  const float* as2 = (const float*)d_in[7];
  const float* ad2 = (const float*)d_in[8];
  const float* b2  = (const float*)d_in[9];
  const float* Wfc = (const float*)d_in[10];
  const float* bfc = (const float*)d_in[11];
  float* out = (float*)d_out;

  const int N = N_NODES;
  const int E = in_sizes[1] / 2;
  const int ET = E + N;

  // ---- workspace layout (zero-init region first, one fill kernel covers it) ----
  char* ws = (char*)d_ws;
  unsigned* emax1 = (unsigned*)ws;  ws += (size_t)N * 4 * 4;
  float*    denom1 = (float*)ws;    ws += (size_t)N * 4 * 4;
  float*    out1   = (float*)ws;    ws += (size_t)N * F1 * 4;
  unsigned* emax2  = (unsigned*)ws; ws += (size_t)N * 4;
  float*    denom2 = (float*)ws;    ws += (size_t)N * 4;
  float*    out2   = (float*)ws;    ws += (size_t)N * HID * 4;
  const long zero_words = (long)N * (4 + 4 + F1 + 1 + 1 + HID);
  float*    h1     = (float*)ws;    ws += (size_t)N * F1 * 4;   // bf16 hb aliases h1 (dead after agg1)
  __bf16*   hb     = (__bf16*)h1;
  float*    a_s1   = (float*)ws;    ws += (size_t)N * 4 * 4;
  float*    a_d1   = (float*)ws;    ws += (size_t)N * 4 * 4;
  float*    a_s2   = (float*)ws;    ws += (size_t)N * 4;
  float*    a_d2   = (float*)ws;    ws += (size_t)N * 4;
  float*    h2     = (float*)ws;    ws += (size_t)N * HID * 4;
  __bf16*   w2t    = (__bf16*)ws;   ws += (size_t)F1 * HID * 2;

  k_fill0<<<(int)((zero_words + 255) / 256), 256, 0, stream>>>((unsigned*)d_ws, zero_words);
  k_gemm1_alpha<<<(N + 7) / 8, 256, 0, stream>>>(x, W1, as1, ad1, h1, a_s1, a_d1);
  k_w2t<<<(F1 * HID + 255) / 256, 256, 0, stream>>>(W2, w2t);
  k_edge_max1<<<(ET + 255) / 256, 256, 0, stream>>>(ei, a_s1, a_d1, emax1, E, ET);
  k_edge_agg1<<<(ET + 7) / 8, 256, 0, stream>>>(ei, a_s1, a_d1, emax1, denom1, h1, out1, E, ET);
  k_norm1<<<(int)(((long)N * F1 + 255) / 256), 256, 0, stream>>>(out1, denom1, b1, hb);
  k_gemm2_wmma<<<(N / 16 + 3) / 4, 128, 0, stream>>>(hb, w2t, h2);
  k_alpha2<<<(N + 7) / 8, 256, 0, stream>>>(h2, as2, ad2, a_s2, a_d2);
  k_edge_max2<<<(ET + 255) / 256, 256, 0, stream>>>(ei, a_s2, a_d2, emax2, E, ET);
  k_edge_agg2<<<(ET + 7) / 8, 256, 0, stream>>>(ei, a_s2, a_d2, emax2, denom2, h2, out2, E, ET);
  k_final<<<(N + 7) / 8, 256, 0, stream>>>(out2, denom2, b2, Wfc, bfc, out);
}
